// Seq2Seq_30846455120651
// MI455X (gfx1250) — matrix operand order repaired and checked by reading, not compiled, not run on hardware
//
#include <hip/hip_runtime.h>

// ---------------------------------------------------------------------------
// CDNA5 (gfx1250) seq2seq forward.  wave32, WMMA bf16 16x16x32, f32 accum.
// ---------------------------------------------------------------------------

typedef __bf16 bhalf;
typedef __attribute__((ext_vector_type(16))) __bf16 bf16x16;
typedef __attribute__((ext_vector_type(8)))  __bf16 bf16x8;
typedef __attribute__((ext_vector_type(8)))  float  f32x8;

#define B_    16
#define S_    128
#define T_    48
#define VOUT_ 32000

__device__ __forceinline__ float sigm_f(float x){ return 1.0f/(1.0f+__expf(-x)); }
__device__ __forceinline__ float selu_f(float x){
  const float a=1.6732632423543772f, s=1.0507009873554805f;
  return s*(x>0.0f ? x : a*(__expf(x)-1.0f));
}

union AFrag { bf16x16 v; bf16x8 h[2]; };

// A tile 16(M) x 32(K), 16-bit layout (ISA 7.12.2):
//   lanes 0-15 : row m=lane, elems 0..7 = K 0..7,  elems 8..15 = K 16..23
//   lanes 16-31: row m=lane-16, elems 0..7 = K 8..15, elems 8..15 = K 24..31
__device__ __forceinline__ bf16x16 load_a16x32(const bhalf* a, int ld, int lane){
  int m  = lane & 15;
  int kb = (lane & 16) ? 8 : 0;
  const bhalf* p = a + (size_t)m*ld + kb;
  AFrag f;
  f.h[0] = *(const bf16x8*)(p);
  f.h[1] = *(const bf16x8*)(p + 16);
  return f.v;
}

// B tile 32(K) x 16(N) taken from row-major W[N,K] (computing A @ W^T):
//   lanes 0-15 : col n=lane,   elems 0..7 = K 0..7,  elems 8..15 = K 16..23
//   lanes 16-31: col n=lane-16, elems 0..7 = K 8..15, elems 8..15 = K 24..31
__device__ __forceinline__ bf16x16 load_b32x16(const bhalf* w, int ld, int lane){
  int n  = lane & 15;
  int kb = (lane & 16) ? 8 : 0;
  const bhalf* p = w + (size_t)n*ld + kb;
  AFrag f;
  f.h[0] = *(const bf16x8*)(p);
  f.h[1] = *(const bf16x8*)(p + 16);
  return f.v;
}

__device__ __forceinline__ f32x8 wmma_bf16(bf16x16 a, bf16x16 b, f32x8 c){
  return __builtin_amdgcn_wmma_f32_16x16x32_bf16(false, a, false, b, (short)0, c, false, false);
}

// ---------------------------------------------------------------------------
// Weight / data prep kernels
// ---------------------------------------------------------------------------

__global__ void k_cast_rows(bhalf* __restrict__ dst, const float* __restrict__ src,
                            int n, int cols, int dld, int doff)
{
  int i = blockIdx.x*256 + threadIdx.x;
  if (i >= n) return;
  int r = i / cols, cc = i - r*cols;
  dst[(size_t)r*dld + doff + cc] = (bhalf)src[i];
}

// x1[(s*16+b)*512+e] = bf16(emb[src[b,s], e])   (time-major [S,B,E])
__global__ void k_embed(bhalf* __restrict__ dst, const float* __restrict__ emb,
                        const int* __restrict__ src, int S)
{
  int i = blockIdx.x*256 + threadIdx.x;
  int n = S*16*512;
  if (i >= n) return;
  int e = i & 511; int b = (i >> 9) & 15; int s = i >> 13;
  dst[i] = (bhalf)emb[(size_t)src[b*S + s]*512 + e];
}

// next-layer input: [S2,B,1024]  row(t2,b) = [hf(2t2) hb(2t2) hf(2t2+1) hb(2t2+1)]
__global__ void k_pack_stride2(bhalf* __restrict__ dst, const bhalf* __restrict__ hsf,
                               const bhalf* __restrict__ hsb, int S2)
{
  int i = blockIdx.x*256 + threadIdx.x;
  int n = S2*16*1024;
  if (i >= n) return;
  int c = i & 1023; int b = (i >> 10) & 15; int t2 = i >> 14;
  int half = c >> 9; int cc = c & 511; int st = 2*t2 + half;
  dst[i] = (cc < 256) ? hsf[((size_t)st*16 + b)*256 + cc]
                      : hsb[((size_t)st*16 + b)*256 + (cc - 256)];
}

// [S3,B,512] row(t,b) = [hf(t) hb(t)]
__global__ void k_pack_concat(bhalf* __restrict__ dst, const bhalf* __restrict__ hsf,
                              const bhalf* __restrict__ hsb, int S3)
{
  int i = blockIdx.x*256 + threadIdx.x;
  int n = S3*16*512;
  if (i >= n) return;
  int c = i & 511; int b = (i >> 9) & 15; int t = i >> 13;
  dst[i] = (c < 256) ? hsf[((size_t)t*16 + b)*256 + c]
                     : hsb[((size_t)t*16 + b)*256 + (c - 256)];
}

__global__ void k_zero_t0(float* __restrict__ out)
{
  int i = blockIdx.x*256 + threadIdx.x;
  if (i >= 16*VOUT_) return;
  int b = i / VOUT_, v = i - b*VOUT_;
  out[(size_t)b*T_*VOUT_ + v] = 0.0f;
}

// ---------------------------------------------------------------------------
// Generic WMMA GEMM:  C[M,N] = epi( A[M,K] @ W[N,K]^T + bias[N] )
//   one 16x16 output tile per wave; EPI 0 = identity (f32 out), 1 = SELU (bf16)
// ---------------------------------------------------------------------------

template<int EPI, typename OutT>
__global__ __launch_bounds__(256) void k_wmma_gemm(
    OutT* __restrict__ C, const bhalf* __restrict__ A, const bhalf* __restrict__ W,
    const float* __restrict__ bias, int M, int N, int K)
{
  int lane = threadIdx.x & 31;
  int wave = blockIdx.x * 8 + (threadIdx.x >> 5);
  int ntiles = N >> 4;
  int total  = (M >> 4) * ntiles;
  if (wave >= total) return;                       // wave-uniform
  int mt = wave / ntiles, nt = wave - mt*ntiles;
  const bhalf* a0 = A + (size_t)(mt << 4) * K;
  const bhalf* w0 = W + (size_t)(nt << 4) * K;
  int n  = (nt << 4) + (lane & 15);
  f32x8 acc; float bv = bias[n];
  #pragma unroll
  for (int e = 0; e < 8; ++e) acc[e] = bv;
  for (int k0 = 0; k0 < K; k0 += 32)
    acc = wmma_bf16(load_a16x32(a0 + k0, K, lane), load_b32x16(w0 + k0, K, lane), acc);
  int mb = (lane & 16) ? 8 : 0;
  #pragma unroll
  for (int e = 0; e < 8; ++e){
    float v = acc[e];
    if (EPI == 1) v = selu_f(v);
    C[(size_t)((mt << 4) + mb + e) * N + n] = (OutT)v;
  }
}

// Logits GEMM: A = H2[47*16, 512] (row = t*16+b), W = fc_w, out[b, t+1, v].
__global__ __launch_bounds__(256) void k_wmma_logits(
    float* __restrict__ C, const bhalf* __restrict__ A, const bhalf* __restrict__ W,
    const float* __restrict__ bias, int Mt, int N, int K, int T)
{
  int lane = threadIdx.x & 31;
  int wave = blockIdx.x * 8 + (threadIdx.x >> 5);
  int ntiles = N >> 4;
  int total  = Mt * ntiles;
  if (wave >= total) return;
  int mt = wave / ntiles, nt = wave - mt*ntiles;
  const bhalf* a0 = A + (size_t)(mt << 4) * K;
  const bhalf* w0 = W + (size_t)(nt << 4) * K;
  int n = (nt << 4) + (lane & 15);
  f32x8 acc; float bv = bias[n];
  #pragma unroll
  for (int e = 0; e < 8; ++e) acc[e] = bv;
  for (int k0 = 0; k0 < K; k0 += 32){
    __builtin_prefetch(w0 + (size_t)(lane & 15) * K + k0 + 64, 0, 1);  // stream fc_w from L2
    acc = wmma_bf16(load_a16x32(a0 + k0, K, lane), load_b32x16(w0 + k0, K, lane), acc);
  }
  int mb = (lane & 16) ? 8 : 0;
  #pragma unroll
  for (int e = 0; e < 8; ++e){
    int b = mb + e;
    C[((size_t)b * T + (mt + 1)) * N + n] = acc[e];
  }
}

// ---------------------------------------------------------------------------
// Persistent encoder LSTM recurrence (one direction per launch, one workgroup).
// hid=256, B=16.  G[t] = x_t @ w_ih^T + b  precomputed (f32, [S,16,1024]).
// Per step: g = G[t] + h @ w_hh^T via WMMA (64 N-tiles, K=256), then gates.
// ---------------------------------------------------------------------------

__global__ __launch_bounds__(1024) void k_lstm_rec(
    bhalf* __restrict__ hs, float* __restrict__ hfin,
    const float* __restrict__ G, const bhalf* __restrict__ Whh, int S, int rev)
{
  extern __shared__ char smem[];
  bhalf* h = (bhalf*)smem;                       // 16*256 bf16   (8 KB)
  float* c = (float*)(smem + 8192);              // 16*256 f32   (16 KB)
  float* g = (float*)(smem + 8192 + 16384);      // 16*1024 f32  (64 KB)
  int tid = threadIdx.x, lane = tid & 31, wave = tid >> 5;  // 32 waves

  for (int i = tid; i < 16*256; i += 1024){ h[i] = (bhalf)0.0f; c[i] = 0.0f; }
  __syncthreads();

  for (int s = 0; s < S; ++s){
    int t = rev ? (S - 1 - s) : s;
    const float* Gt = G + (size_t)t * 16 * 1024;
    #pragma unroll
    for (int q = 0; q < 2; ++q){
      int nt = wave * 2 + q;                     // 64 tiles over 32 waves
      int n  = (nt << 4) + (lane & 15);
      int mb = (lane & 16) ? 8 : 0;
      f32x8 acc;
      #pragma unroll
      for (int e = 0; e < 8; ++e) acc[e] = Gt[(size_t)(mb + e) * 1024 + n];
      const bhalf* w0 = Whh + (size_t)(nt << 4) * 256;
      #pragma unroll
      for (int k0 = 0; k0 < 256; k0 += 32)
        acc = wmma_bf16(load_a16x32(h + k0, 256, lane), load_b32x16(w0 + k0, 256, lane), acc);
      #pragma unroll
      for (int e = 0; e < 8; ++e) g[(mb + e) * 1024 + n] = acc[e];
    }
    __syncthreads();
    for (int i = tid; i < 16*256; i += 1024){
      int b = i >> 8, j = i & 255;
      float gi = g[b*1024 + j],       gf = g[b*1024 + 256 + j];
      float gc = g[b*1024 + 512 + j], go = g[b*1024 + 768 + j];
      float cn = sigm_f(gf)*c[i] + sigm_f(gi)*tanhf(gc);
      float hn = sigm_f(go)*tanhf(cn);
      c[i] = cn; h[i] = (bhalf)hn;
      hs[((size_t)t*16 + b)*256 + j] = (bhalf)hn;
    }
    __syncthreads();
  }
  if (hfin){
    for (int i = tid; i < 16*256; i += 1024){
      int b = i >> 8, j = i & 255;
      hfin[b*512 + j] = (float)h[i];
    }
  }
}

// ---------------------------------------------------------------------------
// Persistent decoder: attention + 2 LSTM cells per step; all state in LDS.
// Writes h2 per step; logits done afterwards as one big GEMM (teacher forcing).
// Wd1 = [d1.w_ih | d1.w_hh] -> [2048,1536];  Wd2 = [d2.w_ih | d2.w_hh] -> [2048,1024]
// ---------------------------------------------------------------------------

__global__ __launch_bounds__(512) void k_decoder(
    const bhalf* __restrict__ keys, const bhalf* __restrict__ vals,
    const float* __restrict__ ench, const int* __restrict__ tgt,
    const int* __restrict__ lens, const bhalf* __restrict__ fcw,
    const bhalf* __restrict__ Wd1, const float* __restrict__ b1,
    const bhalf* __restrict__ Wd2, const float* __restrict__ b2,
    bhalf* __restrict__ H2, int Tm1, int T)
{
  extern __shared__ char smem[];
  bhalf* xc  = (bhalf*)smem;                 // 16*1536: [embed | context | h1]  (48 KB)
  bhalf* h2s = (bhalf*)(smem + 49152);       // 16*512                           (16 KB)
  float* c1  = (float*)(smem + 65536);       // 16*512                           (32 KB)
  float* c2  = (float*)(smem + 98304);       // 16*512                           (32 KB)
  float* g   = (float*)(smem + 131072);      // 16*2048                         (128 KB)
  bhalf* h2a = (bhalf*)(smem + 262144);      // 16*512                           (16 KB)
  float* att = (float*)(smem + 278528);      // 16*32                             (2 KB)
  int tid = threadIdx.x, lane = tid & 31, wave = tid >> 5;   // 16 waves

  for (int i = tid; i < 16*512; i += 512){
    int b = i >> 9, j = i & 511;
    xc[b*1536 + 1024 + j] = (bhalf)0.0f;     // h1 = 0
    h2s[i] = (bhalf)0.0f; c1[i] = 0.0f; c2[i] = 0.0f;
    h2a[i] = (bhalf)ench[i];                 // first attention query = enc_hidden
  }
  __syncthreads();

  for (int t = 0; t < Tm1; ++t){
    { // energies: one (b,s) per thread
      int b = tid >> 5, s = tid & 31;
      const bhalf* kp = keys + ((size_t)s*16 + b)*512;
      const bhalf* q  = h2a + b*512;
      float e = 0.0f;
      #pragma unroll 8
      for (int d = 0; d < 512; ++d) e += (float)kp[d] * (float)q[d];
      att[b*32 + s] = e;
    }
    __syncthreads();
    if (tid < 16){ // softmax, mask, L1-renormalize (reference semantics)
      int L4 = lens[tid] >> 2;
      float mx = -3.4e38f;
      for (int s = 0; s < 32; ++s) mx = fmaxf(mx, att[tid*32+s]);
      float sum = 0.0f;
      for (int s = 0; s < 32; ++s){ float e = __expf(att[tid*32+s]-mx); att[tid*32+s]=e; sum+=e; }
      float inv = 1.0f/sum, ms = 0.0f;
      for (int s = 0; s < 32; ++s){
        float p = att[tid*32+s]*inv * ((s < L4) ? 1.0f : 0.0f);
        att[tid*32+s] = p; ms += p;
      }
      float r = 1.0f/fmaxf(ms, 1e-12f);
      for (int s = 0; s < 32; ++s) att[tid*32+s] *= r;
    }
    __syncthreads();
    for (int i = tid; i < 16*512; i += 512){   // context = attn @ values
      int b = i >> 9, d = i & 511;
      const bhalf* vp = vals + (size_t)b*512 + d;
      float a = 0.0f;
      #pragma unroll 8
      for (int s2 = 0; s2 < 32; ++s2) a += att[b*32+s2] * (float)vp[(size_t)s2*16*512];
      xc[b*1536 + 512 + d] = (bhalf)a;
    }
    for (int i = tid; i < 16*512; i += 512){   // tied embedding = fc_w[word]
      int b = i >> 9, d = i & 511;
      xc[b*1536 + d] = fcw[(size_t)tgt[b*T + t]*512 + d];
    }
    __syncthreads();

    // LSTM1: g = [x|ctx|h1] @ Wd1^T + b1   (N=2048, K=1536)
    for (int q2 = 0; q2 < 8; ++q2){
      int nt = wave*8 + q2;
      int n  = (nt << 4) + (lane & 15);
      int mb = (lane & 16) ? 8 : 0;
      f32x8 acc; float bv = b1[n];
      #pragma unroll
      for (int e = 0; e < 8; ++e) acc[e] = bv;
      const bhalf* w0 = Wd1 + (size_t)(nt << 4) * 1536;
      for (int k0 = 0; k0 < 1536; k0 += 32)
        acc = wmma_bf16(load_a16x32(xc + k0, 1536, lane), load_b32x16(w0 + k0, 1536, lane), acc);
      #pragma unroll
      for (int e = 0; e < 8; ++e) g[(mb + e)*2048 + n] = acc[e];
    }
    __syncthreads();
    for (int i = tid; i < 16*512; i += 512){   // gates1 -> h1 (into xc), c1
      int b = i >> 9, j = i & 511;
      float gi=g[b*2048+j], gf=g[b*2048+512+j], gc=g[b*2048+1024+j], go=g[b*2048+1536+j];
      float cn = sigm_f(gf)*c1[i] + sigm_f(gi)*tanhf(gc);
      float hn = sigm_f(go)*tanhf(cn);
      c1[i] = cn; xc[b*1536 + 1024 + j] = (bhalf)hn;
    }
    __syncthreads();

    // LSTM2: g = [h1|h2s] @ Wd2^T + b2   (N=2048, K=1024, A split across 2 LDS regions)
    for (int q2 = 0; q2 < 8; ++q2){
      int nt = wave*8 + q2;
      int n  = (nt << 4) + (lane & 15);
      int mb = (lane & 16) ? 8 : 0;
      f32x8 acc; float bv = b2[n];
      #pragma unroll
      for (int e = 0; e < 8; ++e) acc[e] = bv;
      const bhalf* w0 = Wd2 + (size_t)(nt << 4) * 1024;
      for (int k0 = 0; k0 < 512; k0 += 32)
        acc = wmma_bf16(load_a16x32(xc + 1024 + k0, 1536, lane), load_b32x16(w0 + k0, 1024, lane), acc);
      for (int k0 = 0; k0 < 512; k0 += 32)
        acc = wmma_bf16(load_a16x32(h2s + k0, 512, lane), load_b32x16(w0 + 512 + k0, 1024, lane), acc);
      #pragma unroll
      for (int e = 0; e < 8; ++e) g[(mb + e)*2048 + n] = acc[e];
    }
    __syncthreads();
    for (int i = tid; i < 16*512; i += 512){   // gates2 -> h2, c2, query update
      int b = i >> 9, j = i & 511;
      float gi=g[b*2048+j], gf=g[b*2048+512+j], gc=g[b*2048+1024+j], go=g[b*2048+1536+j];
      float cn = sigm_f(gf)*c2[i] + sigm_f(gi)*tanhf(gc);
      float hn = sigm_f(go)*tanhf(cn);
      c2[i] = cn;
      bhalf hb = (bhalf)hn;
      h2s[i] = hb; h2a[i] = hb;
      H2[((size_t)t*16 + b)*512 + j] = hb;
    }
    __syncthreads();
  }
}

// ---------------------------------------------------------------------------
// Host orchestration
// ---------------------------------------------------------------------------

extern "C" void kernel_launch(void* const* d_in, const int* in_sizes, int n_in,
                              void* d_out, int out_size, void* d_ws, size_t ws_size,
                              hipStream_t stream)
{
  (void)in_sizes; (void)n_in; (void)out_size; (void)ws_size;
  // setup_inputs() flattened in dict-insertion order:
  const int*   src  = (const int*)d_in[0];
  const int*   lens = (const int*)d_in[1];
  const int*   tgt  = (const int*)d_in[2];
  const float* emb  = (const float*)d_in[3];
  const float* wih1f=(const float*)d_in[4],  *whh1f=(const float*)d_in[5],  *b1f=(const float*)d_in[6];
  const float* wih1b=(const float*)d_in[7],  *whh1b=(const float*)d_in[8],  *b1b=(const float*)d_in[9];
  const float* wih2f=(const float*)d_in[10], *whh2f=(const float*)d_in[11], *b2f=(const float*)d_in[12];
  const float* wih2b=(const float*)d_in[13], *whh2b=(const float*)d_in[14], *b2b=(const float*)d_in[15];
  const float* wih3f=(const float*)d_in[16], *whh3f=(const float*)d_in[17], *b3f=(const float*)d_in[18];
  const float* wih3b=(const float*)d_in[19], *whh3b=(const float*)d_in[20], *b3b=(const float*)d_in[21];
  const float* fc1w =(const float*)d_in[22], *fc1b =(const float*)d_in[23];
  const float* fc2w =(const float*)d_in[24], *fc2b =(const float*)d_in[25];
  const float* fcw  =(const float*)d_in[26], *fcb  =(const float*)d_in[27];
  const float* d1wih=(const float*)d_in[28], *d1whh=(const float*)d_in[29], *d1b=(const float*)d_in[30];
  const float* d2wih=(const float*)d_in[31], *d2whh=(const float*)d_in[32], *d2b=(const float*)d_in[33];
  float* out = (float*)d_out;

  char* wsb = (char*)d_ws; size_t off = 0;
  auto alloc = [&](size_t bytes)->char*{
    char* p = wsb + off; off = (off + bytes + 255) & ~(size_t)255; return p;
  };

  // bf16 weights
  bhalf* Wih1f=(bhalf*)alloc(1024*512*2),  *Whh1f=(bhalf*)alloc(1024*256*2);
  bhalf* Wih1b=(bhalf*)alloc(1024*512*2),  *Whh1b=(bhalf*)alloc(1024*256*2);
  bhalf* Wih2f=(bhalf*)alloc(1024*1024*2), *Whh2f=(bhalf*)alloc(1024*256*2);
  bhalf* Wih2b=(bhalf*)alloc(1024*1024*2), *Whh2b=(bhalf*)alloc(1024*256*2);
  bhalf* Wih3f=(bhalf*)alloc(1024*1024*2), *Whh3f=(bhalf*)alloc(1024*256*2);
  bhalf* Wih3b=(bhalf*)alloc(1024*1024*2), *Whh3b=(bhalf*)alloc(1024*256*2);
  bhalf* Fc1w =(bhalf*)alloc(512*512*2),   *Fc2w =(bhalf*)alloc(512*512*2);
  bhalf* Fcw  =(bhalf*)alloc((size_t)VOUT_*512*2);
  bhalf* Wd1  =(bhalf*)alloc(2048*1536*2), *Wd2  =(bhalf*)alloc(2048*1024*2);
  // activations
  bhalf* X1  =(bhalf*)alloc(2048*512*2);
  float* Gf  =(float*)alloc(2048*1024*4), *Gb =(float*)alloc(2048*1024*4);
  bhalf* Hsf =(bhalf*)alloc(2048*256*2),  *Hsb=(bhalf*)alloc(2048*256*2);
  bhalf* X2  =(bhalf*)alloc(1024*1024*2);
  bhalf* X3  =(bhalf*)alloc(512*1024*2);
  bhalf* X3c =(bhalf*)alloc(512*512*2);
  bhalf* Keys=(bhalf*)alloc(512*512*2),   *Vals=(bhalf*)alloc(512*512*2);
  float* Ench=(float*)alloc(16*512*4);
  bhalf* H2  =(bhalf*)alloc((size_t)47*16*512*2);

  auto castw = [&](bhalf* dst, const float* s, int rows, int cols, int dld, int doff){
    int n = rows*cols;
    k_cast_rows<<<dim3((n+255)/256), dim3(256), 0, stream>>>(dst, s, n, cols, dld, doff);
  };
  auto gemmF32 = [&](float* C, const bhalf* A, const bhalf* W, const float* bias,
                     int M, int N, int K){
    int tiles = (M>>4)*(N>>4);
    k_wmma_gemm<0,float><<<dim3((tiles+7)/8), dim3(256), 0, stream>>>(C, A, W, bias, M, N, K);
  };
  auto gemmSelu = [&](bhalf* C, const bhalf* A, const bhalf* W, const float* bias,
                      int M, int N, int K){
    int tiles = (M>>4)*(N>>4);
    k_wmma_gemm<1,bhalf><<<dim3((tiles+7)/8), dim3(256), 0, stream>>>(C, A, W, bias, M, N, K);
  };

  // ---- weight prep (once per launch; deterministic) ----
  castw(Wih1f, wih1f, 1024,512, 512,0);   castw(Whh1f, whh1f, 1024,256, 256,0);
  castw(Wih1b, wih1b, 1024,512, 512,0);   castw(Whh1b, whh1b, 1024,256, 256,0);
  castw(Wih2f, wih2f, 1024,1024,1024,0);  castw(Whh2f, whh2f, 1024,256, 256,0);
  castw(Wih2b, wih2b, 1024,1024,1024,0);  castw(Whh2b, whh2b, 1024,256, 256,0);
  castw(Wih3f, wih3f, 1024,1024,1024,0);  castw(Whh3f, whh3f, 1024,256, 256,0);
  castw(Wih3b, wih3b, 1024,1024,1024,0);  castw(Whh3b, whh3b, 1024,256, 256,0);
  castw(Fc1w, fc1w, 512,512, 512,0);      castw(Fc2w, fc2w, 512,512, 512,0);
  castw(Fcw,  fcw,  VOUT_,512, 512,0);
  castw(Wd1, d1wih, 2048,1024, 1536,0);   castw(Wd1, d1whh, 2048,512, 1536,1024);
  castw(Wd2, d2wih, 2048,512, 1024,0);    castw(Wd2, d2whh, 2048,512, 1024,512);

  // ---- encoder ----
  k_embed<<<dim3((S_*16*512+255)/256), dim3(256), 0, stream>>>(X1, emb, src, S_);

  const size_t recLds = 8192 + 16384 + 65536;  // 88 KB
  // layer 1 (S=128, Din=512)
  gemmF32(Gf, X1, Wih1f, b1f, 2048, 1024, 512);
  gemmF32(Gb, X1, Wih1b, b1b, 2048, 1024, 512);
  k_lstm_rec<<<dim3(1), dim3(1024), recLds, stream>>>(Hsf, (float*)nullptr, Gf, Whh1f, 128, 0);
  k_lstm_rec<<<dim3(1), dim3(1024), recLds, stream>>>(Hsb, (float*)nullptr, Gb, Whh1b, 128, 1);
  k_pack_stride2<<<dim3((64*16*1024+255)/256), dim3(256), 0, stream>>>(X2, Hsf, Hsb, 64);
  // layer 2 (S=64, Din=1024)
  gemmF32(Gf, X2, Wih2f, b2f, 1024, 1024, 1024);
  gemmF32(Gb, X2, Wih2b, b2b, 1024, 1024, 1024);
  k_lstm_rec<<<dim3(1), dim3(1024), recLds, stream>>>(Hsf, (float*)nullptr, Gf, Whh2f, 64, 0);
  k_lstm_rec<<<dim3(1), dim3(1024), recLds, stream>>>(Hsb, (float*)nullptr, Gb, Whh2b, 64, 1);
  k_pack_stride2<<<dim3((32*16*1024+255)/256), dim3(256), 0, stream>>>(X3, Hsf, Hsb, 32);
  // layer 3 (S=32, Din=1024) + final hiddens
  gemmF32(Gf, X3, Wih3f, b3f, 512, 1024, 1024);
  gemmF32(Gb, X3, Wih3b, b3b, 512, 1024, 1024);
  k_lstm_rec<<<dim3(1), dim3(1024), recLds, stream>>>(Hsf, Ench,       Gf, Whh3f, 32, 0);
  k_lstm_rec<<<dim3(1), dim3(1024), recLds, stream>>>(Hsb, Ench + 256, Gb, Whh3b, 32, 1);
  k_pack_concat<<<dim3((32*16*512+255)/256), dim3(256), 0, stream>>>(X3c, Hsf, Hsb, 32);

  gemmSelu(Keys, X3c, Fc1w, fc1b, 512, 512, 512);
  gemmSelu(Vals, X3c, Fc2w, fc2b, 512, 512, 512);

  // ---- decoder recurrence (persistent, WMMA in LDS) ----
  const size_t decLds = 49152 + 16384 + 32768 + 32768 + 131072 + 16384 + 2048;  // ~274 KB
  k_decoder<<<dim3(1), dim3(512), decLds, stream>>>(Keys, Vals, Ench, tgt, lens, Fcw,
                                                    Wd1, d1b, Wd2, d2b, H2, T_-1, T_);

  // ---- output: pred[:,0,:]=0 ; pred[:,t+1,:] = H2[t] @ fc_w^T + fc_b ----
  k_zero_t0<<<dim3((16*VOUT_+255)/256), dim3(256), 0, stream>>>(out);
  {
    int tiles = 47 * (VOUT_>>4);
    k_wmma_logits<<<dim3((tiles+7)/8), dim3(256), 0, stream>>>(out, H2, Fcw, fcb,
                                                               47, VOUT_, 512, T_);
  }
}
